// MultiHeadAttentionWithBias_75634374082756
// MI455X (gfx1250) — matrix operand order repaired and checked by reading, not compiled, not run on hardware
//
#include <hip/hip_runtime.h>
#include <hip/hip_bf16.h>
#include <math.h>

// ---------------------------------------------------------------------------
// MultiHeadAttentionWithBias for MI455X (gfx1250, wave32, WMMA)
// B=2 L=2048 H=16 Dm=1024 Dk=Dv=64
// proj(Q,K,V) -> flash attention (bias+mask, online softmax) -> out proj.
// GEMMs: v_wmma_f32_16x16x32_bf16.  Attention K/V staging: double-buffered
// global_load_async_to_lds_b128 (ASYNCcnt) + bias/mask global_prefetch.
// ---------------------------------------------------------------------------

typedef __bf16 bf16;
typedef __attribute__((ext_vector_type(8)))  __bf16 bf16x8;
typedef __attribute__((ext_vector_type(16))) __bf16 v16bf;
typedef __attribute__((ext_vector_type(8)))  float  v8f;

#define LSEQ   2048
#define DMODEL 1024
#define NHEAD  16
#define DHEAD  64

// assemble v16bf fragment from two contiguous 8-element runs
__device__ __forceinline__ v16bf make_frag(bf16x8 lo, bf16x8 hi) {
  v16bf r;
#pragma unroll
  for (int i = 0; i < 8; ++i) { r[i] = lo[i]; r[i + 8] = hi[i]; }
  return r;
}

// A-operand (16xK=32, bf16): lane l -> row (l&15); K runs [(l>>4)*8,+8) and
// [16+(l>>4)*8,+8).  base points at (row0,k0), row-major with `stride` elems.
__device__ __forceinline__ v16bf load_A(const bf16* base, int stride, int lane) {
  const int m = lane & 15, kh = lane >> 4;
  const bf16* row = base + m * stride;
  bf16x8 lo = *(const bf16x8*)(row + kh * 8);
  bf16x8 hi = *(const bf16x8*)(row + 16 + kh * 8);
  return make_frag(lo, hi);
}

// B-operand (K=32 x 16, bf16): lane l -> column (l&15); K runs [(l>>4)*8,+8) and
// [16+(l>>4)*8,+8).
// base layout: [n][k] with `stride` elems per n-row (k contiguous).
__device__ __forceinline__ v16bf load_B(const bf16* base, int stride, int lane) {
  const int n = lane & 15, kh = lane >> 4;
  const bf16* row = base + n * stride;
  bf16x8 lo = *(const bf16x8*)(row + kh * 8);
  bf16x8 hi = *(const bf16x8*)(row + 16 + kh * 8);
  return make_frag(lo, hi);
}

__device__ __forceinline__ v8f wmma_bf16(v16bf a, v16bf b, v8f c) {
  // (neg_a, A, neg_b, B, c_mod, C, reuse_a, reuse_b)
  return __builtin_amdgcn_wmma_f32_16x16x32_bf16(false, a, false, b,
                                                 (short)0, c, false, false);
}

// per-lane async global->LDS copy of 16 bytes; tracked by ASYNCcnt
__device__ __forceinline__ void async_b128(const bf16* lds_ptr, const bf16* g) {
  const unsigned lds_byte = (unsigned)(uintptr_t)lds_ptr;  // LDS offset = low 32b
  asm volatile("global_load_async_to_lds_b128 %0, %1, off"
               :: "v"(lds_byte), "v"(g) : "memory");
}
__device__ __forceinline__ void wait_async_le2() {
  asm volatile("s_wait_asynccnt 0x2" ::: "memory");
}

// ---------------------------------------------------------------------------
// Generic GEMM: Y = X(4096 x 1024) * W(1024 x 1024), K-step 32 through LDS.
// Block: 256 thr = 8 waves; wave computes 16 tokens x 64 cols (4 C tiles).
// MODE 0: out bf16 [B,H,L,64]   (Q/K head-split)
// MODE 1: out bf16 [B,H,64,L]   (V head-split, transposed)
// MODE 2: X is bf16 (attn out), out fp32 [token,1024]  (final projection)
// ---------------------------------------------------------------------------
template <int MODE>
__global__ __launch_bounds__(256)
void gemm_bf16_kernel(const float* __restrict__ Xf, const bf16* __restrict__ Xb,
                      const float* __restrict__ W,
                      bf16* __restrict__ outB, float* __restrict__ outF,
                      float scale) {
  __shared__ __align__(16) bf16 sX[128 * 32];   // 128 tokens x 32 k
  __shared__ __align__(16) bf16 sW[64 * 32];    // 64 n x 32 k (transposed)

  const int tid  = threadIdx.x;
  const int lane = tid & 31;
  const int wave = tid >> 5;
  const int tok0 = blockIdx.x * 128;
  const int n0   = blockIdx.y * 64;

  v8f acc[4];
#pragma unroll
  for (int t = 0; t < 4; ++t)
#pragma unroll
    for (int i = 0; i < 8; ++i) acc[t][i] = 0.0f;

  for (int k0 = 0; k0 < DMODEL; k0 += 32) {
    __syncthreads();
    // ---- stage X tile (convert fp32->bf16 on the fly), 16 elems/thread ----
#pragma unroll
    for (int i = 0; i < 2; ++i) {
      const int e = tid * 16 + i * 8;          // 0..4095
      const int r = e >> 5, c = e & 31;
      bf16x8 v;
      if (MODE != 2) {
        const float* xs = Xf + (size_t)(tok0 + r) * DMODEL + k0 + c;
        float4 a = *(const float4*)xs;
        float4 b = *(const float4*)(xs + 4);
        v[0] = (bf16)a.x; v[1] = (bf16)a.y; v[2] = (bf16)a.z; v[3] = (bf16)a.w;
        v[4] = (bf16)b.x; v[5] = (bf16)b.y; v[6] = (bf16)b.z; v[7] = (bf16)b.w;
      } else {
        v = *(const bf16x8*)(Xb + (size_t)(tok0 + r) * DMODEL + k0 + c);
      }
      *(bf16x8*)(sX + e) = v;
    }
    // ---- stage W tile transposed: sW[n*32 + k] ----
    {
      const int kk = tid >> 3;                 // 0..31
      const int nb = (tid & 7) * 8;            // 0..56
      const float* wsrc = W + (size_t)(k0 + kk) * DMODEL + n0 + nb;
      float4 a = *(const float4*)wsrc;
      float4 b = *(const float4*)(wsrc + 4);
      sW[(nb + 0) * 32 + kk] = (bf16)a.x;
      sW[(nb + 1) * 32 + kk] = (bf16)a.y;
      sW[(nb + 2) * 32 + kk] = (bf16)a.z;
      sW[(nb + 3) * 32 + kk] = (bf16)a.w;
      sW[(nb + 4) * 32 + kk] = (bf16)b.x;
      sW[(nb + 5) * 32 + kk] = (bf16)b.y;
      sW[(nb + 6) * 32 + kk] = (bf16)b.z;
      sW[(nb + 7) * 32 + kk] = (bf16)b.w;
    }
    __syncthreads();

    v16bf afrag = load_A(sX + wave * 16 * 32, 32, lane);
#pragma unroll
    for (int t = 0; t < 4; ++t) {
      v16bf bfrag = load_B(sW + t * 16 * 32, 32, lane);
      acc[t] = wmma_bf16(afrag, bfrag, acc[t]);
    }
  }

  // ---- write out (C layout: reg r -> m = r + 8*half, n = lane&15) ----
  const int l15 = lane & 15, half = lane >> 4;
#pragma unroll
  for (int t = 0; t < 4; ++t) {
#pragma unroll
    for (int r = 0; r < 8; ++r) {
      const int token = tok0 + wave * 16 + r + 8 * half;
      const int col   = n0 + t * 16 + l15;
      const float val = acc[t][r] * scale;
      if (MODE == 2) {
        outF[(size_t)token * DMODEL + col] = val;
      } else {
        const int b = token >> 11, tl = token & (LSEQ - 1);
        const int h = col >> 6,  d  = col & (DHEAD - 1);
        if (MODE == 0) {   // [B,H,L,64]
          outB[(((size_t)b * NHEAD + h) * LSEQ + tl) * DHEAD + d] = (bf16)val;
        } else {           // [B,H,64,L]
          outB[(((size_t)b * NHEAD + h) * DHEAD + d) * LSEQ + tl] = (bf16)val;
        }
      }
    }
  }
}

// ---------------------------------------------------------------------------
// Flash attention with bias + mask.
// Grid (L/128, H, B). Block 256 = 8 waves; wave owns 16 q rows x full 64 d.
// K/V tiles double-buffered in LDS via global_load_async_to_lds_b128.
// Per 32-key tile: S = Q*K^T (4 wmma), bias/mask, online softmax (shfl over
// 16-lane halves), P relayout via per-wave LDS, O += P*V (4 wmma).
// ---------------------------------------------------------------------------
__global__ __launch_bounds__(256)
void attn_kernel(const bf16* __restrict__ Qh, const bf16* __restrict__ Kh,
                 const bf16* __restrict__ Vt, const float* __restrict__ bias,
                 const int* __restrict__ mask, bf16* __restrict__ attnOut) {
  __shared__ __align__(16) bf16 sK[2][32 * 64];      // 32 keys x 64 d
  __shared__ __align__(16) bf16 sV[2][64 * 32];      // 64 d x 32 keys (V^T)
  __shared__ __align__(16) bf16 sP[8 * 16 * 32];     // per-wave P tiles

  const int tid  = threadIdx.x;
  const int lane = tid & 31;
  const int wave = tid >> 5;
  const int bh   = blockIdx.z * NHEAD + blockIdx.y;  // b*16 + h
  const int b    = bh >> 4;
  const int h    = bh & 15;
  const int qBase = blockIdx.x * 128 + wave * 16;
  const int l15 = lane & 15, half = lane >> 4;

  // per-thread staging coordinates (one b128 each for K and V tiles)
  const int kr = tid >> 3, kc = (tid & 7) * 8;       // K: 32 x 64
  const int vr = tid >> 2, vc = (tid & 3) * 8;       // V^T: 64 x 32
  const bf16* Kbase = Kh + (size_t)bh * LSEQ * DHEAD;
  const bf16* Vbase = Vt + (size_t)bh * DHEAD * LSEQ;

  // Q fragments for both 32-wide d chunks, held in registers for whole loop
  const bf16* Qrow = Qh + ((size_t)bh * LSEQ + qBase) * DHEAD;
  v16bf qf0 = load_A(Qrow, DHEAD, lane);
  v16bf qf1 = load_A(Qrow + 32, DHEAD, lane);

  v8f o[4];
#pragma unroll
  for (int t = 0; t < 4; ++t)
#pragma unroll
    for (int i = 0; i < 8; ++i) o[t][i] = 0.0f;
  float mrow[8], lrow[8];
#pragma unroll
  for (int r = 0; r < 8; ++r) { mrow[r] = -1.0e30f; lrow[r] = 0.0f; }

  // prologue: async-stage tile 0 into buffer 0
  async_b128(&sK[0][kr * 64 + kc], Kbase + (size_t)kr * DHEAD + kc);
  async_b128(&sV[0][vr * 32 + vc], Vbase + (size_t)vr * LSEQ + vc);

  for (int it = 0; it < LSEQ / 32; ++it) {
    const int kk0 = it * 32;
    const int cur = it & 1;
    const int nxt = cur ^ 1;
    const int kkn = (kk0 + 32 < LSEQ) ? (kk0 + 32) : kk0;  // clamped prefetch

    __syncthreads();  // all waves done reading buf[nxt] (from iteration it-1)
    // issue next tile's async copies into the other buffer
    async_b128(&sK[nxt][kr * 64 + kc],
               Kbase + (size_t)(kkn + kr) * DHEAD + kc);
    async_b128(&sV[nxt][vr * 32 + vc],
               Vbase + (size_t)vr * LSEQ + kkn + vc);
    wait_async_le2();   // in-order completion => current tile has landed
    __syncthreads();    // current tile visible to all waves

    // prefetch next tile's bias/mask rows (32 lanes cover 16 rows x 2 halves)
    {
      const size_t nidx =
          ((size_t)b * LSEQ + qBase + l15) * LSEQ + kkn + half * 16;
      __builtin_prefetch((const void*)(bias + nidx), 0, 1);
      __builtin_prefetch((const void*)(mask + nidx), 0, 1);
    }

    // ---- S = Q * K^T : two 16x16 C tiles over 32 keys ----
    v8f s[2];
#pragma unroll
    for (int t = 0; t < 2; ++t) {
      v8f a;
#pragma unroll
      for (int i = 0; i < 8; ++i) a[i] = 0.0f;
      a = wmma_bf16(qf0, load_B(sK[cur] + t * 16 * 64, 64, lane), a);       // d 0..31
      a = wmma_bf16(qf1, load_B(sK[cur] + t * 16 * 64 + 32, 64, lane), a);  // d 32..63
      s[t] = a;
    }

    // ---- bias + mask (1/sqrt(d) already folded into Q projection) ----
#pragma unroll
    for (int t = 0; t < 2; ++t) {
#pragma unroll
      for (int r = 0; r < 8; ++r) {
        const int qrow = qBase + r + 8 * half;
        const int key  = kk0 + t * 16 + l15;
        const size_t idx = ((size_t)b * LSEQ + qrow) * LSEQ + key;
        float sv = s[t][r] + bias[idx];
        s[t][r] = (mask[idx] == 0) ? -10000.0f : sv;
      }
    }

    // ---- online softmax: rows live within a 16-lane half ----
    float alpha[8];
#pragma unroll
    for (int r = 0; r < 8; ++r) {
      float mx = fmaxf(s[0][r], s[1][r]);
#pragma unroll
      for (int off = 1; off < 16; off <<= 1)
        mx = fmaxf(mx, __shfl_xor(mx, off, 32));
      const float mnew = fmaxf(mrow[r], mx);
      alpha[r] = __expf(mrow[r] - mnew);
      mrow[r] = mnew;
      const float p0 = __expf(s[0][r] - mnew);
      const float p1 = __expf(s[1][r] - mnew);
      s[0][r] = p0; s[1][r] = p1;
      float sum = p0 + p1;
#pragma unroll
      for (int off = 1; off < 16; off <<= 1)
        sum += __shfl_xor(sum, off, 32);
      lrow[r] = lrow[r] * alpha[r] + sum;
    }
#pragma unroll
    for (int t = 0; t < 4; ++t)
#pragma unroll
      for (int r = 0; r < 8; ++r) o[t][r] *= alpha[r];

    // ---- P: C layout -> A layout through per-wave LDS tile ----
    bf16* pbuf = sP + wave * (16 * 32);
#pragma unroll
    for (int t = 0; t < 2; ++t)
#pragma unroll
      for (int r = 0; r < 8; ++r)
        pbuf[(r + 8 * half) * 32 + t * 16 + l15] = (bf16)s[t][r];
    // same-wave DS ordering is in-order; compiler inserts s_wait_dscnt
    v16bf pfrag = load_A(pbuf, 32, lane);

    // ---- O += P * V ----
#pragma unroll
    for (int t = 0; t < 4; ++t) {
      v16bf vfrag = load_B(sV[cur] + t * 16 * 32, 32, lane);
      o[t] = wmma_bf16(pfrag, vfrag, o[t]);
    }
  }

  // ---- normalize and emit attn output [token, H*64] as bf16 ----
  float inv[8];
#pragma unroll
  for (int r = 0; r < 8; ++r) inv[r] = 1.0f / lrow[r];
#pragma unroll
  for (int t = 0; t < 4; ++t) {
#pragma unroll
    for (int r = 0; r < 8; ++r) {
      const int qrow = qBase + r + 8 * half;
      const size_t tok = (size_t)b * LSEQ + qrow;
      attnOut[tok * DMODEL + h * DHEAD + t * 16 + l15] = (bf16)(o[t][r] * inv[r]);
    }
  }
}

// ---------------------------------------------------------------------------
extern "C" void kernel_launch(void* const* d_in, const int* in_sizes, int n_in,
                              void* d_out, int out_size, void* d_ws, size_t ws_size,
                              hipStream_t stream) {
  const float* q    = (const float*)d_in[0];
  const float* k    = (const float*)d_in[1];
  const float* v    = (const float*)d_in[2];
  const int*   mask = (const int*)  d_in[3];
  const float* bias = (const float*)d_in[4];
  const float* w_qs = (const float*)d_in[5];
  const float* w_ks = (const float*)d_in[6];
  const float* w_vs = (const float*)d_in[7];
  const float* w_fc = (const float*)d_in[8];
  float* out = (float*)d_out;

  // workspace: 4 bf16 buffers of 4096*1024 elements (8 MB each)
  bf16* ws   = (bf16*)d_ws;
  const size_t SEG = (size_t)4096 * 1024;
  bf16* Qh   = ws;            // [B,H,L,64]
  bf16* Kh   = ws + SEG;      // [B,H,L,64]
  bf16* Vt   = ws + 2 * SEG;  // [B,H,64,L]
  bf16* attn = ws + 3 * SEG;  // [token, H*64]

  const dim3 gGemm(32, 16), bGemm(256);
  const float invTemp = 0.125f;  // 1/sqrt(64), folded into Q

  gemm_bf16_kernel<0><<<gGemm, bGemm, 0, stream>>>(q, nullptr, w_qs, Qh, nullptr, invTemp);
  gemm_bf16_kernel<0><<<gGemm, bGemm, 0, stream>>>(k, nullptr, w_ks, Kh, nullptr, 1.0f);
  gemm_bf16_kernel<1><<<gGemm, bGemm, 0, stream>>>(v, nullptr, w_vs, Vt, nullptr, 1.0f);

  attn_kernel<<<dim3(LSEQ / 128, NHEAD, 2), dim3(256), 0, stream>>>(
      Qh, Kh, Vt, bias, mask, attn);

  gemm_bf16_kernel<2><<<gGemm, bGemm, 0, stream>>>(nullptr, attn, w_fc, nullptr, out, 1.0f);
}
